// LWSA_35098472743469
// MI455X (gfx1250) — matrix operand order repaired and checked by reading, not compiled, not run on hardware
//
#include <hip/hip_runtime.h>
#include <hip/hip_bf16.h>

typedef __attribute__((ext_vector_type(16))) _Float16 v16h;
typedef __attribute__((ext_vector_type(8)))  _Float16 v8h;
typedef __attribute__((ext_vector_type(8)))  float    v8f;

#define B_   2
#define C_   256
#define N_   1728          // 12*12*12 voxels
#define L_   6912          // N_ * 4 heads folded into tokens
#define DQK_ 32
#define DV_  64
#define QBLK 108           // L_/64 query-tile groups per batch
#define NB_  108           // L_/64 key blocks

__device__ __forceinline__ v16h pack16(const _Float16* lo, const _Float16* hi) {
    v8h a = *(const v8h*)lo;
    v8h b = *(const v8h*)hi;
    v16h r;
#pragma unroll
    for (int i = 0; i < 8; ++i) { r[i] = a[i]; r[i + 8] = b[i]; }
    return r;
}

// 16-lane (DPP row) max butterfly: xor1, xor2, xor4, xor8 equivalents.
// Pure VALU — keeps the LDS pipe free for the WMMA operand streams.
__device__ __forceinline__ float rowmax16(float x) {
    x = fmaxf(x, __int_as_float(__builtin_amdgcn_update_dpp(
            0, __float_as_int(x), 0xB1, 0xF, 0xF, true)));   // quad_perm [1,0,3,2]
    x = fmaxf(x, __int_as_float(__builtin_amdgcn_update_dpp(
            0, __float_as_int(x), 0x4E, 0xF, 0xF, true)));   // quad_perm [2,3,0,1]
    x = fmaxf(x, __int_as_float(__builtin_amdgcn_update_dpp(
            0, __float_as_int(x), 0x141, 0xF, 0xF, true)));  // row_half_mirror
    x = fmaxf(x, __int_as_float(__builtin_amdgcn_update_dpp(
            0, __float_as_int(x), 0x140, 0xF, 0xF, true)));  // row_mirror
    return x;
}

__device__ __forceinline__ unsigned lds_off(const void* p) {
    return (unsigned)(unsigned long long)p;   // low 32 bits of generic = LDS offset
}

// ---------------------------------------------------------------------------
// Kernel 1: Q/K projections; fold head reshape; fold (1/sqrt(dh))*log2(e)
// into Q so softmax is raw exp2. f16 token-major [b][t][32].
// ---------------------------------------------------------------------------
__global__ __launch_bounds__(256) void lwsa_qk_proj(
    const float* __restrict__ x,  const float* __restrict__ wq,
    const float* __restrict__ bq, const float* __restrict__ wk,
    const float* __restrict__ bk, _Float16* __restrict__ Qh,
    _Float16* __restrict__ Kh)
{
    int idx = blockIdx.x * blockDim.x + threadIdx.x;   // B_*128*N_
    if (idx >= B_ * 128 * N_) return;
    int n = idx % N_;
    int o = (idx / N_) & 127;
    int b = idx / (N_ * 128);

    const float* xb = x + (size_t)b * C_ * N_ + n;
    float sq = 0.f, sk = 0.f;
#pragma unroll 8
    for (int c = 0; c < C_; ++c) {
        float xv = xb[(size_t)c * N_];
        sq += wq[o * C_ + c] * xv;
        sk += wk[o * C_ + c] * xv;
    }
    sq += bq[o];
    sk += bk[o];
    sq *= 0.17677669529663687f * 1.4426950408889634f;  // scale * log2(e)

    int t = n * 4 + (o >> 5);
    int d = o & 31;
    Qh[((size_t)b * L_ + t) * DQK_ + d] = (_Float16)sq;
    Kh[((size_t)b * L_ + t) * DQK_ + d] = (_Float16)sk;
}

// ---------------------------------------------------------------------------
// Kernel 2: repack x into V^T f16, e-major: VhT[b][e][t] = x[b][(t%4)*64+e][t/4]
// ---------------------------------------------------------------------------
__global__ __launch_bounds__(256) void lwsa_v_pack(
    const float* __restrict__ x, _Float16* __restrict__ VhT)
{
    int idx = blockIdx.x * blockDim.x + threadIdx.x;   // B_*C_*N_
    if (idx >= B_ * C_ * N_) return;
    int n = idx % N_;
    int c = (idx / N_) & 255;
    int b = idx / (N_ * C_);
    int e = c & 63, h = c >> 6;
    VhT[((size_t)b * DV_ + e) * L_ + n * 4 + h] = (_Float16)x[idx];
}

// ---------------------------------------------------------------------------
// Kernel 3: flash attention. 16-row query tile per wave, 4 waves/block.
// Double-buffered GLOBAL_LOAD_ASYNC_TO_LDS_B128 staging (ASYNCcnt),
// DPP row-max, rowsum via P*ones WMMA, f16 WMMA 16x16x32 throughout.
// ---------------------------------------------------------------------------
__global__ __launch_bounds__(128) void lwsa_attn(
    const _Float16* __restrict__ Qh, const _Float16* __restrict__ Kh,
    const _Float16* __restrict__ VhT, float* __restrict__ Oacc)
{
    __shared__ __align__(128) _Float16 Ksh[2][64][32];   // [buf][key][d]   4 KB/buf
    __shared__ __align__(128) _Float16 Vsh[2][64][64];   // [buf][e][key]   8 KB/buf
    __shared__ __align__(128) _Float16 Psh[4][16][64];   // per-wave P      8 KB

    const int tid  = threadIdx.x;
    const int wave = tid >> 5;
    const int lane = tid & 31;
    const int nIdx = lane & 15;
    const int half = lane >> 4;

    const int b  = blockIdx.x / QBLK;
    const int q0 = (blockIdx.x % QBLK) * 64 + wave * 16;

    const _Float16* Qb = Qh  + (size_t)b * L_ * DQK_;
    const _Float16* Kb = Kh  + (size_t)b * L_ * DQK_;
    const _Float16* Vb = VhT + (size_t)b * DV_ * L_;

    const unsigned kshB = lds_off(&Ksh[0][0][0]);
    const unsigned vshB = lds_off(&Vsh[0][0][0]);

    // Q tile in A-matrix layout (lane = row m, K split by lane-half)
    const _Float16* qrow = Qb + (size_t)(q0 + nIdx) * DQK_;
    v16h aQ = pack16(qrow + half * 8, qrow + 16 + half * 8);

    v16h vones;
#pragma unroll
    for (int i = 0; i < 16; ++i) vones[i] = (_Float16)1.0f;

    float m_r[8], l_r[8];
    v8f acc[4];
#pragma unroll
    for (int r = 0; r < 8; ++r) { m_r[r] = -__builtin_inff(); l_r[r] = 0.f; }
#pragma unroll
    for (int j = 0; j < 4; ++j) { v8f zz = {}; acc[j] = zz; }

    // stage: issue 6 async b128 loads per thread for one 64-key block
    auto stage = [&](int blk, int buf) {
        const _Float16* ksrc = Kb + (size_t)blk * 64 * DQK_;
        unsigned kd = kshB + (unsigned)buf * 4096u;
#pragma unroll
        for (int it = 0; it < 2; ++it) {
            int c = tid + it * 128;                        // 256 x 16 B = 4 KB
            asm volatile("global_load_async_to_lds_b128 %0, %1, off"
                         :: "v"(kd + (unsigned)c * 16u), "v"(ksrc + c * 8)
                         : "memory");
        }
        const _Float16* vsrc = Vb + blk * 64;
        unsigned vd = vshB + (unsigned)buf * 8192u;
#pragma unroll
        for (int it = 0; it < 4; ++it) {
            int c = tid + it * 128;                        // 512 x 16 B = 8 KB
            int e = c >> 3, part = c & 7;
            asm volatile("global_load_async_to_lds_b128 %0, %1, off"
                         :: "v"(vd + (unsigned)(e * 128 + part * 16)),
                            "v"(vsrc + (size_t)e * L_ + part * 8)
                         : "memory");
        }
    };

    stage(0, 0);                                           // prime the pipeline

    for (int i = 0; i < NB_; ++i) {
        const int cur = i & 1;
        __syncthreads();             // all waves done reading buf[cur^1] (iter i-1)
        if (i + 1 < NB_) {
            stage(i + 1, cur ^ 1);   // prefetch next block while we compute
            asm volatile("s_wait_asynccnt 0x6" ::: "memory");   // block i done
        } else {
            asm volatile("s_wait_asynccnt 0x0" ::: "memory");
        }
        __syncthreads();             // buf[cur] complete across all waves

        // S = Q * K^T : four 16x16 tiles, one v_wmma_f32_16x16x32_f16 each
        v8f s[4];
#pragma unroll
        for (int tk = 0; tk < 4; ++tk) {
            const _Float16* kr = &Ksh[cur][tk * 16 + nIdx][half * 8];
            v16h bK = pack16(kr, kr + 16);
            v8f z = {};
            s[tk] = __builtin_amdgcn_wmma_f32_16x16x32_f16(false, aQ, false, bK,
                                                           (short)0, z, false, false);
        }

        // online softmax: DPP row-max, exp2 (scale*log2e folded into Q)
        float a_r[8];
#pragma unroll
        for (int r = 0; r < 8; ++r) {
            float mx = fmaxf(fmaxf(s[0][r], s[1][r]), fmaxf(s[2][r], s[3][r]));
            mx = rowmax16(mx);
            float mn = fmaxf(m_r[r], mx);
            a_r[r] = exp2f(m_r[r] - mn);
            m_r[r] = mn;
#pragma unroll
            for (int tk = 0; tk < 4; ++tk) s[tk][r] = exp2f(s[tk][r] - mn);
        }

        // C-layout -> row-major f16 P (per-wave LDS scratch)
#pragma unroll
        for (int r = 0; r < 8; ++r) {
            _Float16* pr = &Psh[wave][r + half * 8][0];
#pragma unroll
            for (int tk = 0; tk < 4; ++tk) pr[tk * 16 + nIdx] = (_Float16)s[tk][r];
        }
        __syncthreads();

        // P in A-layout (two K=32 slabs)
        const _Float16* prow = &Psh[wave][nIdx][0];
        v16h aP0 = pack16(prow + half * 8,      prow + 16 + half * 8);
        v16h aP1 = pack16(prow + 32 + half * 8, prow + 48 + half * 8);

        // row sums on the matrix core: rowsum(P) = P * ones(64x16)
        v8f z = {};
        v8f ps = __builtin_amdgcn_wmma_f32_16x16x32_f16(false, aP0, false, vones,
                                                        (short)0, z, false, false);
        ps = __builtin_amdgcn_wmma_f32_16x16x32_f16(false, aP1, false, vones,
                                                    (short)0, ps, false, false);
#pragma unroll
        for (int r = 0; r < 8; ++r) l_r[r] = a_r[r] * l_r[r] + ps[r];

        // O = alpha*O + P*V : 8 WMMAs (4 e-tiles x 2 K-slabs)
#pragma unroll
        for (int j = 0; j < 4; ++j) {
            v8f cacc;
#pragma unroll
            for (int r = 0; r < 8; ++r) cacc[r] = acc[j][r] * a_r[r];
            const _Float16* vr = &Vsh[cur][j * 16 + nIdx][0];
            v16h bV0 = pack16(vr + half * 8,      vr + 16 + half * 8);
            v16h bV1 = pack16(vr + 32 + half * 8, vr + 48 + half * 8);
            cacc   = __builtin_amdgcn_wmma_f32_16x16x32_f16(false, aP0, false, bV0,
                                                            (short)0, cacc, false, false);
            acc[j] = __builtin_amdgcn_wmma_f32_16x16x32_f16(false, aP1, false, bV1,
                                                            (short)0, cacc, false, false);
        }
    }

    // normalize and write O (f32, token-major [b][t][64])
    float* Ob = Oacc + ((size_t)b * L_ + q0) * DV_;
#pragma unroll
    for (int r = 0; r < 8; ++r) {
        int m = r + half * 8;
        float inv = 1.0f / l_r[r];
#pragma unroll
        for (int j = 0; j < 4; ++j)
            Ob[(size_t)m * DV_ + j * 16 + nIdx] = acc[j][r] * inv;
    }
}

// ---------------------------------------------------------------------------
// Kernel 4: grouped 3x3x3 / 5x5x5 convs + tanh skip + attention un-fold
// ---------------------------------------------------------------------------
__global__ __launch_bounds__(256) void lwsa_skip_epilogue(
    const float* __restrict__ x,   const float* __restrict__ ws1,
    const float* __restrict__ bs1, const float* __restrict__ ws2,
    const float* __restrict__ bs2, const float* __restrict__ attn,
    float* __restrict__ out)
{
    int idx = blockIdx.x * blockDim.x + threadIdx.x;   // B_*C_*N_
    if (idx >= B_ * C_ * N_) return;
    int n  = idx % N_;
    int oc = (idx / N_) & 255;
    int b  = idx / (N_ * C_);
    int zz = n / 144, yy = (n / 12) % 12, xx = n % 12;

    float s;
    if (oc < 128) {
        const float* w   = ws1 + oc * 54;               // (2,3,3,3)
        const float* xc0 = x + ((size_t)b * C_ + 2 * oc) * N_;
        s = bs1[oc];
#pragma unroll
        for (int kd = 0; kd < 3; ++kd)
#pragma unroll
            for (int kh = 0; kh < 3; ++kh)
#pragma unroll
                for (int kw = 0; kw < 3; ++kw) {
                    int z = zz + kd - 1, y = yy + kh - 1, xw = xx + kw - 1;
                    if ((unsigned)z < 12u && (unsigned)y < 12u && (unsigned)xw < 12u) {
                        int nn = z * 144 + y * 12 + xw;
                        int wi = kd * 9 + kh * 3 + kw;
                        s += w[wi] * xc0[nn] + w[27 + wi] * xc0[N_ + nn];
                    }
                }
    } else {
        int g = oc - 128;
        const float* w   = ws2 + g * 250;               // (2,5,5,5)
        const float* xc0 = x + ((size_t)b * C_ + 2 * g) * N_;
        s = bs2[g];
        for (int kd = 0; kd < 5; ++kd)
            for (int kh = 0; kh < 5; ++kh)
#pragma unroll
                for (int kw = 0; kw < 5; ++kw) {
                    int z = zz + kd - 2, y = yy + kh - 2, xw = xx + kw - 2;
                    if ((unsigned)z < 12u && (unsigned)y < 12u && (unsigned)xw < 12u) {
                        int nn = z * 144 + y * 12 + xw;
                        int wi = kd * 25 + kh * 5 + kw;
                        s += w[wi] * xc0[nn] + w[125 + wi] * xc0[N_ + nn];
                    }
                }
    }

    int t = n * 4 + (oc >> 6);
    int e = oc & 63;
    out[idx] = attn[((size_t)b * L_ + t) * DV_ + e] + tanhf(s);
}

// ---------------------------------------------------------------------------
extern "C" void kernel_launch(void* const* d_in, const int* in_sizes, int n_in,
                              void* d_out, int out_size, void* d_ws, size_t ws_size,
                              hipStream_t stream) {
    (void)in_sizes; (void)n_in; (void)out_size; (void)ws_size;
    const float* x   = (const float*)d_in[0];
    const float* wq  = (const float*)d_in[1];
    const float* bq  = (const float*)d_in[2];
    const float* wk  = (const float*)d_in[3];
    const float* bk  = (const float*)d_in[4];
    const float* ws1 = (const float*)d_in[5];
    const float* bs1 = (const float*)d_in[6];
    const float* ws2 = (const float*)d_in[7];
    const float* bs2 = (const float*)d_in[8];

    char* ws = (char*)d_ws;
    _Float16* Qh  = (_Float16*)ws;                       //  884736 B
    _Float16* Kh  = Qh + (size_t)B_ * L_ * DQK_;         //  884736 B
    _Float16* VhT = Kh + (size_t)B_ * L_ * DQK_;         // 1769472 B
    float*    Oac = (float*)(ws + 3538944);              // 3538944 B

    lwsa_qk_proj<<<(B_ * 128 * N_ + 255) / 256, 256, 0, stream>>>(
        x, wq, bq, wk, bk, Qh, Kh);
    lwsa_v_pack<<<(B_ * C_ * N_ + 255) / 256, 256, 0, stream>>>(x, VhT);
    lwsa_attn<<<B_ * QBLK, 128, 0, stream>>>(Qh, Kh, VhT, Oac);
    lwsa_skip_epilogue<<<(B_ * C_ * N_ + 255) / 256, 256, 0, stream>>>(
        x, ws1, bs1, ws2, bs2, Oac, (float*)d_out);
}
